// SparseTCAM_22127671509478
// MI455X (gfx1250) — matrix operand order repaired and checked by reading, not compiled, not run on hardware
//
#include <hip/hip_runtime.h>
#include <cstdint>
#include <cstddef>

#define BB 4
#define SS 1024
#define DD 1536
#define HH 24
#define HD 64
#define WINDOW 256
#define ANCHORS 4

typedef __attribute__((ext_vector_type(16))) __bf16 v16bf;
typedef __attribute__((ext_vector_type(8)))  __bf16 v8bf;
typedef __attribute__((ext_vector_type(8)))  float  v8f;
typedef __attribute__((ext_vector_type(4)))  unsigned v4u;
typedef __attribute__((ext_vector_type(8)))  int      v8i_;
typedef __attribute__((ext_vector_type(4)))  int      v4i_;

// ---------------- device helpers ----------------

static __device__ __forceinline__ __bf16 f2bf(float f) {
  unsigned u = __builtin_bit_cast(unsigned, f);
  u += 0x7FFFu + ((u >> 16) & 1u);            // round-to-nearest-even
  unsigned short h = (unsigned short)(u >> 16);
  return __builtin_bit_cast(__bf16, h);
}

static __device__ __forceinline__ float sigmoidf_(float x) {
  return 1.0f / (1.0f + __expf(-x));
}

// A-matrix fragment (16xK row-major source, one K=32 chunk).
// ISA 16-bit A layout: lanes 0-15 hold K {0..7,16..23}, lanes 16-31 hold K {8..15,24..31}.
static __device__ __forceinline__ v16bf loadA16(const __bf16* base, int ld, int lane) {
  int row  = lane & 15;
  int koff = (lane >> 4) << 3;
  const __bf16* p = base + (size_t)row * ld + koff;
  v8bf lo = *(const v8bf*)p;
  v8bf hi = *(const v8bf*)(p + 16);
  v16bf r;
#pragma unroll
  for (int i = 0; i < 8; ++i) { r[i] = lo[i]; r[8 + i] = hi[i]; }
  return r;
}

// B-matrix fragment from an NxK row-major source (one K=32 chunk).
// ISA 16-bit B layout: lane = n, lanes 0-15 hold K {0..7,16..23}, lanes 16-31 hold K {8..15,24..31}.
static __device__ __forceinline__ v16bf loadB16(const __bf16* base, int ld, int lane) {
  int n  = lane & 15;
  int kh = (lane >> 4) << 3;
  const __bf16* p = base + (size_t)n * ld + kh;
  v8bf lo = *(const v8bf*)p;
  v8bf hi = *(const v8bf*)(p + 16);
  v16bf r;
#pragma unroll
  for (int i = 0; i < 8; ++i) { r[i] = lo[i]; r[8 + i] = hi[i]; }
  return r;
}

// ---------------- kernels ----------------

__global__ __launch_bounds__(256) void k_f32_to_bf16(const float* __restrict__ src,
                                                     __bf16* __restrict__ dst, long n) {
  long i = (long)blockIdx.x * blockDim.x + threadIdx.x;
  if (i < n) dst[i] = f2bf(src[i]);
}

// active mask + bf16 operand staging
__global__ __launch_bounds__(256) void k_prep(const float* __restrict__ spikes,
                                              const float* __restrict__ cont,
                                              __bf16* __restrict__ XsB,
                                              __bf16* __restrict__ XcB,
                                              float* __restrict__ activeF) {
  int row = blockIdx.x, t = threadIdx.x;
  const float* sp = spikes + (size_t)row * DD;
  const float* cp = cont   + (size_t)row * DD;
  float s = 0.f;
#pragma unroll
  for (int i = 0; i < 6; ++i) s += sp[t + 256 * i];
  __shared__ float red[256];
  red[t] = s; __syncthreads();
  for (int off = 128; off > 0; off >>= 1) {
    if (t < off) red[t] += red[t + off];
    __syncthreads();
  }
  float act = red[0] > 0.f ? 1.f : 0.f;
  if (t == 0) activeF[row] = act;
  __bf16* xs = XsB + (size_t)row * DD;
  __bf16* xc = XcB + (size_t)row * DD;
#pragma unroll
  for (int i = 0; i < 6; ++i) {
    int e = t + 256 * i;
    xs[e] = f2bf(sp[e] * act);
    xc[e] = f2bf(cp[e]);
  }
}

// C[M,N] f32 = A[M,K] bf16 * W[N,K]^T bf16
// Block = 4 waves sharing one 16-row A tile (staged to LDS via TDM), each wave owns a 16x64 N-tile.
// Grid: (M/16) * (N/256) blocks of 128 threads.
__global__ __launch_bounds__(128) void k_gemm(const __bf16* __restrict__ A,
                                              const __bf16* __restrict__ W,
                                              float* __restrict__ C,
                                              int M, int N, int K) {
  __shared__ __bf16 lds_a[16 * DD];   // 48 KB A tile (K <= DD)
  int wave = threadIdx.x >> 5, lane = threadIdx.x & 31;
  int ntg4 = N >> 8;                  // N / 256
  int mt  = blockIdx.x / ntg4;
  int ntg = blockIdx.x % ntg4;
  int nt  = ntg * 4 + wave;
  const __bf16* Ag = A + (size_t)(mt * 16) * K;

#if __has_builtin(__builtin_amdgcn_tensor_load_to_lds) && __has_builtin(__builtin_amdgcn_s_wait_tensorcnt)
  if (wave == 0) {
    // Tensor DMA descriptor (CDNA5 ISA ch.8): 2-D tile, data_size=2B,
    // tensor/tile = K x 16, dim0 stride = K, type=2 ("image").
    unsigned long long ga = (unsigned long long)(uintptr_t)Ag;
    unsigned ldsOff = (unsigned)(uintptr_t)(&lds_a[0]);   // low 32 bits = LDS byte offset
    v4u g0;
    g0[0] = 1u;                                   // count=1, user descriptor
    g0[1] = ldsOff;                               // lds_addr
    g0[2] = (unsigned)ga;                         // global_addr[31:0]
    g0[3] = (unsigned)((ga >> 32) & 0x1FFFFFFull) | (2u << 30);  // global_addr[56:32] | type=2
    v8i_ g1;
    g1[0] = (1 << 16);                            // data_size = 2 bytes
    g1[1] = (int)((K & 0xFFFF) << 16);            // tensor_dim0[15:0]
    g1[2] = (int)(((unsigned)K >> 16) | (16u << 16));  // tensor_dim0[31:16] | tensor_dim1=16
    g1[3] = (int)((unsigned)(K & 0xFFFF) << 16);  // tile_dim0 = K
    g1[4] = 16;                                   // tile_dim1 = 16, tile_dim2 = 0
    g1[5] = K;                                    // tensor_dim0_stride[31:0]
    g1[6] = 0;
    g1[7] = 0;
    v4i_ gz;
    gz[0] = 0; gz[1] = 0; gz[2] = 0; gz[3] = 0;
    v8i_ gz8;
    gz8[0] = 0; gz8[1] = 0; gz8[2] = 0; gz8[3] = 0;
    gz8[4] = 0; gz8[5] = 0; gz8[6] = 0; gz8[7] = 0;
    // 6-arg form (this toolchain): (g0, g1, g2, g3, extra group, cpol)
    __builtin_amdgcn_tensor_load_to_lds(g0, g1, gz, gz, gz8, 0);
    __builtin_amdgcn_s_wait_tensorcnt(0);
  }
  __syncthreads();
#else
  // fallback: cooperative staging with plain vector loads
  for (int i = threadIdx.x; i < (16 * K) / 8; i += 128)
    *(v8bf*)(&lds_a[i * 8]) = *(const v8bf*)(Ag + (size_t)i * 8);
  __syncthreads();
#endif

  v8f acc0, acc1, acc2, acc3;
#pragma unroll
  for (int i = 0; i < 8; ++i) { acc0[i] = 0.f; acc1[i] = 0.f; acc2[i] = 0.f; acc3[i] = 0.f; }

  const __bf16* Wb = W + (size_t)(nt * 64) * K;
  for (int k0 = 0; k0 < K; k0 += 32) {
    v16bf a  = loadA16(&lds_a[k0], K, lane);      // ds_load from shared A tile
    v16bf b0 = loadB16(Wb + (size_t)(0 * 16) * K + k0, K, lane);
    v16bf b1 = loadB16(Wb + (size_t)(1 * 16) * K + k0, K, lane);
    v16bf b2 = loadB16(Wb + (size_t)(2 * 16) * K + k0, K, lane);
    v16bf b3 = loadB16(Wb + (size_t)(3 * 16) * K + k0, K, lane);
    acc0 = __builtin_amdgcn_wmma_f32_16x16x32_bf16(false, a, false, b0, (short)0, acc0, false, false);
    acc1 = __builtin_amdgcn_wmma_f32_16x16x32_bf16(false, a, false, b1, (short)0, acc1, false, false);
    acc2 = __builtin_amdgcn_wmma_f32_16x16x32_bf16(false, a, false, b2, (short)0, acc2, false, false);
    acc3 = __builtin_amdgcn_wmma_f32_16x16x32_bf16(false, a, false, b3, (short)0, acc3, false, false);
  }
  int half = lane >> 4, col = lane & 15;
  v8f accs[4] = {acc0, acc1, acc2, acc3};
#pragma unroll
  for (int g = 0; g < 4; ++g)
#pragma unroll
    for (int v = 0; v < 8; ++v)
      C[(size_t)(mt * 16 + v + 8 * half) * N + nt * 64 + g * 16 + col] = accs[g][v];
}

// per-(b,d) leaky recurrence over S
__global__ __launch_bounds__(256) void k_decay(const float* __restrict__ tcam,
                                               const float* __restrict__ decay_raw,
                                               float* __restrict__ outD) {
  int tid = blockIdx.x * blockDim.x + threadIdx.x;
  if (tid >= BB * DD) return;
  int b = tid / DD, d = tid % DD;
  float a = sigmoidf_(decay_raw[d >> 6]);
  const float* src = tcam + (size_t)b * SS * DD + d;
  float*       dst = outD + (size_t)b * SS * DD + d;
  float h = 0.f;
  for (int s = 0; s < SS; ++s) {
    h = a * h + src[(size_t)s * DD];
    dst[(size_t)s * DD] = h;
  }
}

// mode 0: RoPE, write bf16 [b,h,s,hd]; mode 1: plain transpose to bf16 [b,h,hd,s]
__global__ __launch_bounds__(256) void k_rope_tr(const float* __restrict__ X,
                                                 __bf16* __restrict__ Out, int mode) {
  long t = (long)blockIdx.x * blockDim.x + threadIdx.x;
  if (t >= (long)BB * SS * HH * HD) return;
  int hd = (int)(t & 63);
  int h  = (int)((t >> 6) % HH);
  int s  = (int)((t / (64 * HH)) % SS);
  int b  = (int)(t / ((long)64 * HH * SS));
  size_t src = ((size_t)(b * SS + s)) * DD + h * HD + hd;
  float x = X[src];
  if (mode == 0) {
    int hh = hd & 31;
    float invf = __expf(-((float)(2 * hh) * (1.0f / HD)) * 9.210340371976184f); // ln(1e4)
    float sn, cs;
    __sincosf((float)s * invf, &sn, &cs);
    float xr = (hd < 32) ? -X[src + 32] : X[src - 32];
    Out[((size_t)(b * HH + h) * SS + s) * HD + hd] = f2bf(x * cs + xr * sn);
  } else {
    Out[((size_t)(b * HH + h) * HD + hd) * SS + s] = f2bf(x);
  }
}

// flash-style masked attention: one wave per (b,h,16-query tile), 32 keys/step
__global__ __launch_bounds__(128) void k_attn(const __bf16* __restrict__ Qr,
                                              const __bf16* __restrict__ Kr,
                                              const __bf16* __restrict__ Vt,
                                              const float* __restrict__ activeF,
                                              float* __restrict__ attnO) {
  __shared__ float pbuf[4][16 * 33];
  int wave = threadIdx.x >> 5, lane = threadIdx.x & 31;
  int wid = blockIdx.x * 4 + wave;
  int qt = wid & 63;
  int bh = wid >> 6;
  if (bh >= BB * HH) return;
  int h = bh % HH, b = bh / HH;
  int q0 = qt << 4;

  const __bf16* Qb = Qr + (size_t)bh * SS * HD;
  const __bf16* Kb = Kr + (size_t)bh * SS * HD;
  const __bf16* Vb = Vt + (size_t)bh * HD * SS;
  int half = lane >> 4, col = lane & 15, koff = half * 8;

  v16bf qa0 = loadA16(Qb + (size_t)q0 * HD, HD, lane);
  v16bf qa1 = loadA16(Qb + (size_t)q0 * HD + 32, HD, lane);

  v8f acc[4];
#pragma unroll
  for (int g = 0; g < 4; ++g)
#pragma unroll
    for (int i = 0; i < 8; ++i) acc[g][i] = 0.f;
  float rmax[8], rsum[8];
#pragma unroll
  for (int v = 0; v < 8; ++v) { rmax[v] = -3e38f; rsum[v] = 0.f; }
  float* P = pbuf[wave];

  auto doBlock = [&](int kb) {
    v8f s0, s1;
#pragma unroll
    for (int i = 0; i < 8; ++i) { s0[i] = 0.f; s1[i] = 0.f; }
    v16bf b00 = loadB16(Kb + (size_t)kb * HD, HD, lane);
    v16bf b01 = loadB16(Kb + (size_t)kb * HD + 32, HD, lane);
    v16bf b10 = loadB16(Kb + (size_t)(kb + 16) * HD, HD, lane);
    v16bf b11 = loadB16(Kb + (size_t)(kb + 16) * HD + 32, HD, lane);
    s0 = __builtin_amdgcn_wmma_f32_16x16x32_bf16(false, qa0, false, b00, (short)0, s0, false, false);
    s0 = __builtin_amdgcn_wmma_f32_16x16x32_bf16(false, qa1, false, b01, (short)0, s0, false, false);
    s1 = __builtin_amdgcn_wmma_f32_16x16x32_bf16(false, qa0, false, b10, (short)0, s1, false, false);
    s1 = __builtin_amdgcn_wmma_f32_16x16x32_bf16(false, qa1, false, b11, (short)0, s1, false, false);

    int k0i = kb + col, k1i = kb + 16 + col;
    float ak0 = activeF[b * SS + k0i];
    float ak1 = activeF[b * SS + k1i];
#pragma unroll
    for (int v = 0; v < 8; ++v) {
      int q = q0 + v + half * 8;
      int d0 = q - k0i, d1 = q - k1i;
      bool m0 = (d0 >= 0) && ((d0 <= WINDOW) || (k0i < ANCHORS)) && (ak0 > 0.f);
      bool m1 = (d1 >= 0) && ((d1 <= WINDOW) || (k1i < ANCHORS)) && (ak1 > 0.f);
      s0[v] = m0 ? s0[v] * 0.125f : -1e30f;   // HD^-0.5
      s1[v] = m1 ? s1[v] * 0.125f : -1e30f;
    }
#pragma unroll
    for (int v = 0; v < 8; ++v) {
      float tm = fmaxf(s0[v], s1[v]);
      tm = fmaxf(tm, __shfl_xor(tm, 1, 32));
      tm = fmaxf(tm, __shfl_xor(tm, 2, 32));
      tm = fmaxf(tm, __shfl_xor(tm, 4, 32));
      tm = fmaxf(tm, __shfl_xor(tm, 8, 32));
      float nm = fmaxf(rmax[v], tm);
      float p0, p1, sc;
      if (nm < -1e29f) { p0 = 0.f; p1 = 0.f; sc = 1.f; }  // no valid key yet
      else {
        sc = __expf(rmax[v] - nm);
        p0 = __expf(s0[v] - nm);
        p1 = __expf(s1[v] - nm);
        rmax[v] = nm;
      }
      float ts = p0 + p1;
      ts += __shfl_xor(ts, 1, 32);
      ts += __shfl_xor(ts, 2, 32);
      ts += __shfl_xor(ts, 4, 32);
      ts += __shfl_xor(ts, 8, 32);
      rsum[v] = rsum[v] * sc + ts;
#pragma unroll
      for (int g = 0; g < 4; ++g) acc[g][v] *= sc;
      P[(half * 8 + v) * 33 + col]      = p0;
      P[(half * 8 + v) * 33 + col + 16] = p1;
    }
    asm volatile("s_wait_dscnt 0" ::: "memory");
    v16bf pa;  // re-swizzle C layout -> A layout via LDS
#pragma unroll
    for (int i = 0; i < 8; ++i) {
      pa[i]     = f2bf(P[col * 33 + koff + i]);
      pa[8 + i] = f2bf(P[col * 33 + koff + 16 + i]);
    }
#pragma unroll
    for (int g = 0; g < 4; ++g) {
      v16bf vb = loadB16(Vb + (size_t)(g * 16) * SS + kb, SS, lane);
      acc[g] = __builtin_amdgcn_wmma_f32_16x16x32_bf16(
          false, pa, false, vb, (short)0, acc[g], false, false);
    }
  };

  int lo = q0 - WINDOW; if (lo < 0) lo = 0;
  int kb_lo = lo & ~31;
  if (kb_lo > 0) doBlock(0);                 // anchor block
  for (int kb = kb_lo; kb <= q0; kb += 32) doBlock(kb);

#pragma unroll
  for (int v = 0; v < 8; ++v) {
    float inv = rsum[v] > 0.f ? 1.0f / rsum[v] : 0.f;   // has_key guard
    int q = q0 + v + half * 8;
    size_t base = ((size_t)(b * SS + q)) * DD + h * HD + col;
#pragma unroll
    for (int g = 0; g < 4; ++g) attnO[base + g * 16] = acc[g][v] * inv;
  }
}

__global__ __launch_bounds__(256) void k_mix(const float* __restrict__ decayO,
                                             const float* __restrict__ attnO,
                                             const float* __restrict__ activeF,
                                             const float* __restrict__ attn_gate,
                                             __bf16* __restrict__ mixB) {
  long i = (long)blockIdx.x * blockDim.x + threadIdx.x;
  if (i >= (long)BB * SS * DD) return;
  long row = i / DD;
  float g = sigmoidf_(attn_gate[0]);
  float a = attnO[i] * activeF[row];
  mixB[i] = f2bf((1.f - g) * decayO[i] + g * a);
}

__global__ __launch_bounds__(256) void k_lnorm(const float* __restrict__ X,
                                               const float* __restrict__ gamma,
                                               const float* __restrict__ beta,
                                               float* __restrict__ out,
                                               float* __restrict__ spk) {
  int row = blockIdx.x, t = threadIdx.x;
  const float* x = X + (size_t)row * DD;
  float s = 0.f, s2 = 0.f, buf[6];
#pragma unroll
  for (int i = 0; i < 6; ++i) {
    float v = x[t + 256 * i];
    buf[i] = v; s += v; s2 += v * v;
  }
  __shared__ float r1[256], r2[256];
  r1[t] = s; r2[t] = s2; __syncthreads();
  for (int off = 128; off > 0; off >>= 1) {
    if (t < off) { r1[t] += r1[t + off]; r2[t] += r2[t + off]; }
    __syncthreads();
  }
  float mu   = r1[0] * (1.f / DD);
  float var  = r2[0] * (1.f / DD) - mu * mu;
  float rstd = rsqrtf(var + 1e-5f);
#pragma unroll
  for (int i = 0; i < 6; ++i) {
    int e = t + 256 * i;
    float y = (buf[i] - mu) * rstd * gamma[e] + beta[e];
    out[(size_t)row * DD + e] = y;
    spk[(size_t)row * DD + e] = (y >= 1.0f) ? 1.f : 0.f;
  }
}

// ---------------- launcher ----------------

extern "C" void kernel_launch(void* const* d_in, const int* in_sizes, int n_in,
                              void* d_out, int out_size, void* d_ws, size_t ws_size,
                              hipStream_t stream) {
  (void)in_sizes; (void)n_in; (void)out_size; (void)ws_size;
  const float* spikes    = (const float*)d_in[0];
  const float* cont      = (const float*)d_in[1];
  const float* Wt        = (const float*)d_in[2];
  const float* Wo        = (const float*)d_in[3];
  const float* gamma     = (const float*)d_in[4];
  const float* beta      = (const float*)d_in[5];
  const float* decay_raw = (const float*)d_in[6];
  const float* Wq        = (const float*)d_in[7];
  const float* Wk        = (const float*)d_in[8];
  const float* Wv        = (const float*)d_in[9];
  const float* attn_gate = (const float*)d_in[10];

  const size_t NTOK = (size_t)BB * SS;        // 4096
  const size_t NELT = NTOK * DD;              // 6,291,456
  const size_t WELT = (size_t)DD * DD;

  char* ws = (char*)d_ws;
  size_t off = 0;
  auto alloc = [&](size_t bytes) -> char* {
    char* p = ws + off;
    off += (bytes + 4096 + 4095) & ~(size_t)4095;   // +>=4KB guard for tail-tile overreads
    return p;
  };
  float*  activeF = (float*)alloc(NTOK * 4);
  __bf16* XsB     = (__bf16*)alloc(NELT * 2);
  __bf16* XcB     = (__bf16*)alloc(NELT * 2);
  __bf16* WtB     = (__bf16*)alloc(WELT * 2);
  __bf16* WqB     = (__bf16*)alloc(WELT * 2);
  __bf16* WkB     = (__bf16*)alloc(WELT * 2);
  __bf16* WvB     = (__bf16*)alloc(WELT * 2);
  __bf16* WoB     = (__bf16*)alloc(WELT * 2);
  float*  tcamF   = (float*)alloc(NELT * 4);
  float*  QF      = (float*)alloc(NELT * 4);
  float*  KF      = (float*)alloc(NELT * 4);
  float*  VF      = (float*)alloc(NELT * 4);
  float*  decayF  = (float*)alloc(NELT * 4);
  __bf16* QrB     = (__bf16*)alloc(NELT * 2);
  __bf16* KrB     = (__bf16*)alloc(NELT * 2);
  __bf16* VtB     = (__bf16*)alloc(NELT * 2);
  // aliases with disjoint lifetimes
  float*  attnF   = QF;    // Q f32 dead after rope
  float*  outPreF = KF;    // K f32 dead after rope
  __bf16* mixB    = XsB;   // Xs dead after tcam GEMM

  // 1. weights -> bf16
  {
    long n = (long)WELT; int grid = (int)((n + 255) / 256);
    k_f32_to_bf16<<<grid, 256, 0, stream>>>(Wt, WtB, n);
    k_f32_to_bf16<<<grid, 256, 0, stream>>>(Wq, WqB, n);
    k_f32_to_bf16<<<grid, 256, 0, stream>>>(Wk, WkB, n);
    k_f32_to_bf16<<<grid, 256, 0, stream>>>(Wv, WvB, n);
    k_f32_to_bf16<<<grid, 256, 0, stream>>>(Wo, WoB, n);
  }
  // 2. active mask + bf16 operands
  k_prep<<<(int)NTOK, 256, 0, stream>>>(spikes, cont, XsB, XcB, activeF);

  // 3. projection GEMMs (TDM-staged A tile, 4 waves per block)
  const int M = BB * SS, N = DD, K = DD;
  const int gblocks = (M / 16) * (N / 256);   // 256 * 6 = 1536
  k_gemm<<<gblocks, 128, 0, stream>>>(XsB, WtB, tcamF, M, N, K);
  k_gemm<<<gblocks, 128, 0, stream>>>(XcB, WqB, QF,    M, N, K);
  k_gemm<<<gblocks, 128, 0, stream>>>(XcB, WkB, KF,    M, N, K);
  k_gemm<<<gblocks, 128, 0, stream>>>(XcB, WvB, VF,    M, N, K);

  // 4. leaky recurrence
  k_decay<<<(BB * DD + 255) / 256, 256, 0, stream>>>(tcamF, decay_raw, decayF);

  // 5. RoPE + layout transforms
  {
    long n = (long)NELT; int grid = (int)((n + 255) / 256);
    k_rope_tr<<<grid, 256, 0, stream>>>(QF, QrB, 0);
    k_rope_tr<<<grid, 256, 0, stream>>>(KF, KrB, 0);
    k_rope_tr<<<grid, 256, 0, stream>>>(VF, VtB, 1);
  }

  // 6. masked flash attention
  k_attn<<<(BB * HH * (SS / 16)) / 4, 128, 0, stream>>>(QrB, KrB, VtB, activeF, attnF);

  // 7. gated blend -> bf16
  {
    long n = (long)NELT; int grid = (int)((n + 255) / 256);
    k_mix<<<grid, 256, 0, stream>>>(decayF, attnF, activeF, attn_gate, mixB);
  }

  // 8. output projection
  k_gemm<<<gblocks, 128, 0, stream>>>(mixB, WoB, outPreF, M, N, K);

  // 9. layernorm + surrogate spike -> d_out (out first, then spikes_out)
  float* outF = (float*)d_out;
  k_lnorm<<<(int)NTOK, 256, 0, stream>>>(outPreF, gamma, beta, outF, outF + NELT);
}
